// LocalAttention_10282151707573
// MI455X (gfx1250) — compile-verified, matrix operand order checked by reading, not run on hardware
//
#include <hip/hip_runtime.h>

// ---------------------------------------------------------------------------
// Local banded attention (WINDOW=128), B=2, S=2048, E=1024, H=16, D=64.
// bf16 WMMA (v_wmma_f32_16x16x32_bf16, wave32) for all 4 GEMMs + attention.
// GEMM: weights pre-transposed at conversion; weight tiles staged into a
// double-buffered LDS tile with GLOBAL_LOAD_ASYNC_TO_LDS_B128 (ASYNCcnt),
// 16 WMMAs per wave per k-tile.
// ---------------------------------------------------------------------------

typedef __attribute__((ext_vector_type(16))) __bf16          v16bf;
typedef __attribute__((ext_vector_type(16))) unsigned short  v16u;
typedef __attribute__((ext_vector_type(8)))  float           v8f;

#define B_  2
#define S_  2048
#define E_  1024
#define H_  16
#define D_  64
#define W_  128

struct U16x16 { uint4 lo, hi; };

static __device__ __forceinline__ unsigned short f2bf(float f) {
    unsigned int u = __float_as_uint(f);
    unsigned int r = u + 0x7FFFu + ((u >> 16) & 1u);   // round-to-nearest-even
    return (unsigned short)(r >> 16);
}
static __device__ __forceinline__ v16bf bc(v16u x) {
    return __builtin_bit_cast(v16bf, x);
}
static __device__ __forceinline__ v16u pack2(uint4 a, uint4 b) {
    U16x16 t{a, b};
    return __builtin_bit_cast(v16u, t);
}
static __device__ __forceinline__ v8f zero8() {
    v8f r;
    #pragma unroll
    for (int i = 0; i < 8; ++i) r[i] = 0.f;
    return r;
}
static __device__ __forceinline__ float rmax16(float x) {
    #pragma unroll
    for (int m = 1; m < 16; m <<= 1) x = fmaxf(x, __shfl_xor(x, m, 16));
    return x;
}
static __device__ __forceinline__ float rsum16(float x) {
    #pragma unroll
    for (int m = 1; m < 16; m <<= 1) x += __shfl_xor(x, m, 16);
    return x;
}

// ---------------------------------------------------------------------------
// f32 -> bf16 (plain)
// ---------------------------------------------------------------------------
__global__ void cvt_bf16(const float* __restrict__ src,
                         unsigned short* __restrict__ dst, int n) {
    int i = blockIdx.x * blockDim.x + threadIdx.x;
    int stride = gridDim.x * blockDim.x;
    for (; i < n; i += stride) dst[i] = f2bf(src[i]);
}

// ---------------------------------------------------------------------------
// f32 [K][N] -> bf16 transposed [N][K] via LDS 32x32 tile.
// ---------------------------------------------------------------------------
__global__ void __launch_bounds__(256)
cvt_tr_bf16(const float* __restrict__ W, unsigned short* __restrict__ Wt) {
    __shared__ unsigned short t[32][33];
    const int kt = blockIdx.x * 32, nt = blockIdx.y * 32;
    const int tx = threadIdx.x & 31, ty = threadIdx.x >> 5;   // ty in [0,8)
    #pragma unroll
    for (int i = 0; i < 4; ++i) {
        int r = ty + i * 8;
        t[tx][r] = f2bf(W[(size_t)(kt + r) * E_ + nt + tx]);
    }
    __syncthreads();
    #pragma unroll
    for (int i = 0; i < 4; ++i) {
        int r = ty + i * 8;
        Wt[(size_t)(nt + r) * E_ + kt + tx] = t[r][tx];
    }
}

// ---------------------------------------------------------------------------
// bf16 GEMM  C[4096,1024] = A[4096,1024] * W + bias, W TRANSPOSED as Wt[n][k].
// Block 256 thr = 8 waves; wave tile 32(M) x 64(N); block tile 256 x 64.
// k-tile 64: async-copied (GLOBAL_LOAD_ASYNC_TO_LDS_B128) into double-
// buffered LDS; s_wait_asynccnt + barrier; 16 WMMAs per wave per tile.
// Fragment layouts per CDNA5 ISA 7.12.2.
// MODE 0: bf16 [B,H,S,D]   MODE 2: bf16 [B,H,D,S]   MODE 3: f32 [M,N]
// ---------------------------------------------------------------------------
template <int MODE>
__global__ void __launch_bounds__(256)
gemm_bf16(const unsigned short* __restrict__ A,
          const unsigned short* __restrict__ Wt,
          const float* __restrict__ bias,
          unsigned short* __restrict__ outb,
          float* __restrict__ outf) {
    constexpr int K = E_;
    // ONLY shared object in this kernel -> LDS offset 0, raw byte offsets
    // are valid VDST values for the async-to-LDS instructions.
    __shared__ unsigned short lb[2][64 * 64];   // [buf][n][k], 8 KB each

    const int tid  = threadIdx.x;
    const int wid  = tid >> 5;
    const int lane = tid & 31;
    const int l15  = lane & 15;
    const int hi   = lane >> 4;
    const int kbb  = hi * 8;
    const int mbase = blockIdx.x * 256;
    const int nbase = blockIdx.y * 64;
    const int mrow0 = mbase + wid * 32 + l15;
    const int mrow1 = mrow0 + 16;

    // per-thread staging slots: 2 x 16B covering the 64x64 ushort tile
    const int e0 = tid, e1 = tid + 256;
    const int n0 = e0 >> 3, ko0 = (e0 & 7) * 8;
    const int n1 = e1 >> 3, ko1 = (e1 & 7) * 8;
    const unsigned short* w0 = Wt + (size_t)(nbase + n0) * K + ko0;
    const unsigned short* w1 = Wt + (size_t)(nbase + n1) * K + ko1;
    const unsigned d0 = (unsigned)((n0 * 64 + ko0) * 2);   // LDS byte offsets
    const unsigned d1 = (unsigned)((n1 * 64 + ko1) * 2);

    // async stage of k-tile `kk` into buffer `buf` (2 instructions / wave)
    auto stage = [&](int buf, int kk) {
        unsigned long long g0 = (unsigned long long)(uintptr_t)(w0 + kk);
        unsigned long long g1 = (unsigned long long)(uintptr_t)(w1 + kk);
        unsigned l0 = d0 + (unsigned)buf * 8192u;
        unsigned l1 = d1 + (unsigned)buf * 8192u;
        asm volatile("global_load_async_to_lds_b128 %0, %1, off"
                     :: "v"(l0), "v"(g0) : "memory");
        asm volatile("global_load_async_to_lds_b128 %0, %1, off"
                     :: "v"(l1), "v"(g1) : "memory");
    };

    v8f acc[2][4];
    #pragma unroll
    for (int r = 0; r < 2; ++r)
        #pragma unroll
        for (int t = 0; t < 4; ++t) acc[r][t] = zero8();

    stage(0, 0);                                   // prologue

    for (int kk = 0; kk < K; kk += 64) {
        const int cur = (kk >> 6) & 1;
        if (kk + 64 < K) {
            stage(cur ^ 1, kk + 64);               // overlap next copy
            __builtin_prefetch(A + (size_t)mrow0 * K + kk + 64, 0, 3);
            asm volatile("s_wait_asynccnt 0x2" ::: "memory");
        } else {
            asm volatile("s_wait_asynccnt 0x0" ::: "memory");
        }
        __syncthreads();                           // tile `cur` visible

        const unsigned short* lc = lb[cur];
        #pragma unroll
        for (int ks = 0; ks < 64; ks += 32) {
            const unsigned short* ap0 = A + (size_t)mrow0 * K + kk + ks + kbb;
            const unsigned short* ap1 = A + (size_t)mrow1 * K + kk + ks + kbb;
            v16u af0 = pack2(*(const uint4*)ap0, *(const uint4*)(ap0 + 16));
            v16u af1 = pack2(*(const uint4*)ap1, *(const uint4*)(ap1 + 16));
            #pragma unroll
            for (int t = 0; t < 4; ++t) {
                const unsigned short* bp = lc + (t * 16 + l15) * 64 + ks + kbb;
                v16u bf_ = pack2(*(const uint4*)bp, *(const uint4*)(bp + 16));
                acc[0][t] = __builtin_amdgcn_wmma_f32_16x16x32_bf16(
                    false, bc(af0), false, bc(bf_), (short)0, acc[0][t], false, false);
                acc[1][t] = __builtin_amdgcn_wmma_f32_16x16x32_bf16(
                    false, bc(af1), false, bc(bf_), (short)0, acc[1][t], false, false);
            }
        }
        __syncthreads();                           // done reading `cur`
    }

    // epilogue: bias + layout-specific store (compile-time MODE)
    #pragma unroll
    for (int r = 0; r < 2; ++r) {
        #pragma unroll
        for (int t = 0; t < 4; ++t) {
            int ng = nbase + t * 16 + l15;
            float bv = bias[ng];
            #pragma unroll
            for (int v = 0; v < 8; ++v) {
                int mg = mbase + wid * 32 + r * 16 + v + 8 * hi;
                float val = acc[r][t][v] + bv;
                if (MODE == 0) {                       // [B,H,S,D]
                    int b = mg >> 11, s = mg & (S_ - 1);
                    int h = ng >> 6,  d = ng & (D_ - 1);
                    outb[((((b << 4) + h) * S_) + s) * D_ + d] = f2bf(val);
                } else if (MODE == 2) {                // [B,H,D,S]
                    int b = mg >> 11, s = mg & (S_ - 1);
                    int h = ng >> 6,  d = ng & (D_ - 1);
                    outb[((((b << 4) + h) * D_) + d) * S_ + s] = f2bf(val);
                } else {                               // f32 [M,N]
                    outf[(size_t)mg * E_ + ng] = val;
                }
            }
        }
    }
}

// ---------------------------------------------------------------------------
// Banded attention: one wave per (b, h, 16-query tile); 32-key chunks,
// online softmax, P re-laid out through per-wave LDS into A-fragments.
// ---------------------------------------------------------------------------
__global__ void __launch_bounds__(128)
attn_banded(const unsigned short* __restrict__ qg,
            const unsigned short* __restrict__ kg,
            const unsigned short* __restrict__ vtg,   // [B,H,D,S]
            unsigned short* __restrict__ ao) {        // [B*S, E] bf16
    __shared__ unsigned short pbuf[4][16 * 32];       // per-wave P tile

    const int tid  = threadIdx.x;
    const int wid  = tid >> 5;
    const int lane = tid & 31;
    const int l15  = lane & 15;
    const int hi   = lane >> 4;
    const int kb   = hi * 8;

    const int w     = blockIdx.x * 4 + wid;           // 0 .. 4095
    const int qt    = w & 127;
    const int h     = (w >> 7) & 15;
    const int b     = w >> 11;
    const int qbase = qt * 16;

    const unsigned short* qh = qg  + (size_t)((b * H_ + h) * S_) * D_;
    const unsigned short* kh = kg  + (size_t)((b * H_ + h) * S_) * D_;
    const unsigned short* vh = vtg + (size_t)((b * H_ + h) * D_) * S_;

    // Q fragments (k = d 0..31 and 32..63), loaded once
    v16u qa0, qa1;
    {
        const unsigned short* qp = qh + (qbase + l15) * D_ + kb;
        qa0 = pack2(*(const uint4*)qp,        *(const uint4*)(qp + 16));
        qa1 = pack2(*(const uint4*)(qp + 32), *(const uint4*)(qp + 48));
    }

    v8f o[4];
    #pragma unroll
    for (int t = 0; t < 4; ++t) o[t] = zero8();
    float rm[8], rs[8];
    #pragma unroll
    for (int v = 0; v < 8; ++v) { rm[v] = -1e30f; rs[v] = 0.f; }

    int j0 = qbase - W_; if (j0 < 0) j0 = 0;
    j0 &= ~31;
    int j1 = qbase + 16 + W_; if (j1 > S_) j1 = S_;
    const float scale = 0.03125f;   // 1/sqrt(1024)

    for (int jc = j0; jc < j1; jc += 32) {
        // scores: q(16xD) @ k^T(Dx32) as two 16x16 tiles, K=64
        v8f sc[2];
        #pragma unroll
        for (int g = 0; g < 2; ++g) {
            int key = jc + g * 16 + l15;
            int kc = key < S_ ? key : S_ - 1;         // clamp; masked below
            const unsigned short* kp = kh + (size_t)kc * D_ + kb;
            v16u kb0 = pack2(*(const uint4*)kp,        *(const uint4*)(kp + 16));
            v16u kb1 = pack2(*(const uint4*)(kp + 32), *(const uint4*)(kp + 48));
            sc[g] = zero8();
            sc[g] = __builtin_amdgcn_wmma_f32_16x16x32_bf16(
                false, bc(qa0), false, bc(kb0), (short)0, sc[g], false, false);
            sc[g] = __builtin_amdgcn_wmma_f32_16x16x32_bf16(
                false, bc(qa1), false, bc(kb1), (short)0, sc[g], false, false);
        }

        // mask + scale + online softmax (per-row stats, 8 rows/lane)
        const int k0 = jc + l15, k1 = jc + 16 + l15;
        #pragma unroll
        for (int v = 0; v < 8; ++v) {
            int qrow = qbase + v + 8 * hi;
            float s0 = sc[0][v] * scale;
            float s1 = sc[1][v] * scale;
            int dd0 = qrow - k0; if (dd0 < 0) dd0 = -dd0;
            int dd1 = qrow - k1; if (dd1 < 0) dd1 = -dd1;
            if (k0 >= S_ || dd0 > W_) s0 = -1e30f;
            if (k1 >= S_ || dd1 > W_) s1 = -1e30f;
            float tmax = rmax16(fmaxf(s0, s1));
            float nm = fmaxf(rm[v], tmax);
            float f  = __expf(rm[v] - nm);
            rm[v] = nm;
            float p0 = __expf(s0 - nm);
            float p1 = __expf(s1 - nm);
            rs[v] = rs[v] * f + rsum16(p0 + p1);
            #pragma unroll
            for (int t = 0; t < 4; ++t) o[t][v] *= f;
            int m = v + 8 * hi;
            pbuf[wid][m * 32 + l15]      = f2bf(p0);
            pbuf[wid][m * 32 + 16 + l15] = f2bf(p1);
        }

        // P as A-fragment (16 queries x 32 keys) via per-wave LDS
        const unsigned short* pp = &pbuf[wid][l15 * 32 + kb];
        v16u pa = pack2(*(const uint4*)pp, *(const uint4*)(pp + 16));

        // O += P @ V  (V^T rows contiguous in S; masked keys carry p=0)
        #pragma unroll
        for (int t = 0; t < 4; ++t) {
            const unsigned short* vp = vh + (size_t)(t * 16 + l15) * S_ + jc + kb;
            v16u vb = pack2(*(const uint4*)vp, *(const uint4*)(vp + 16));
            o[t] = __builtin_amdgcn_wmma_f32_16x16x32_bf16(
                false, bc(pa), false, bc(vb), (short)0, o[t], false, false);
        }
    }

    // normalize + store attn output bf16 [B*S, E]
    #pragma unroll
    for (int t = 0; t < 4; ++t) {
        #pragma unroll
        for (int v = 0; v < 8; ++v) {
            int m = v + 8 * hi;
            float val = o[t][v] / rs[v];
            size_t row = (size_t)(b * S_ + qbase + m);
            ao[row * E_ + h * D_ + t * 16 + l15] = f2bf(val);
        }
    }
}

// ---------------------------------------------------------------------------
// Host launcher
// ---------------------------------------------------------------------------
extern "C" void kernel_launch(void* const* d_in, const int* in_sizes, int n_in,
                              void* d_out, int out_size, void* d_ws, size_t ws_size,
                              hipStream_t stream) {
    const float* x  = (const float*)d_in[0];
    const float* Wq = (const float*)d_in[1];
    const float* bq = (const float*)d_in[2];
    const float* Wk = (const float*)d_in[3];
    const float* bk = (const float*)d_in[4];
    const float* Wv = (const float*)d_in[5];
    const float* bv = (const float*)d_in[6];
    const float* Wo = (const float*)d_in[7];
    const float* bo = (const float*)d_in[8];
    float* out = (float*)d_out;

    const size_t NX = (size_t)B_ * S_ * E_;   // 4M elems
    const size_t NW = (size_t)E_ * E_;        // 1M elems

    char* ws = (char*)d_ws;
    size_t off = 0;
    unsigned short* xb  = (unsigned short*)(ws + off); off += NX * 2;
    unsigned short* wqt = (unsigned short*)(ws + off); off += NW * 2;  // [n][k]
    unsigned short* wkt = (unsigned short*)(ws + off); off += NW * 2;
    unsigned short* wvt = (unsigned short*)(ws + off); off += NW * 2;
    unsigned short* wot = (unsigned short*)(ws + off); off += NW * 2;
    unsigned short* qb  = (unsigned short*)(ws + off); off += NX * 2;  // [B,H,S,D]
    unsigned short* kb  = (unsigned short*)(ws + off); off += NX * 2;  // [B,H,S,D]
    unsigned short* vtb = (unsigned short*)(ws + off); off += NX * 2;  // [B,H,D,S]
    unsigned short* aob = (unsigned short*)(ws + off); off += NX * 2;  // [B*S,E]
    if (off > ws_size) return;   // need 48 MB of workspace

    // conversions: x plain; weights transposed to [n][k]
    cvt_bf16<<<4096, 256, 0, stream>>>(x, xb, (int)NX);
    dim3 tg(E_ / 32, E_ / 32);
    cvt_tr_bf16<<<tg, 256, 0, stream>>>(Wq, wqt);
    cvt_tr_bf16<<<tg, 256, 0, stream>>>(Wk, wkt);
    cvt_tr_bf16<<<tg, 256, 0, stream>>>(Wv, wvt);
    cvt_tr_bf16<<<tg, 256, 0, stream>>>(Wo, wot);

    dim3 gg(B_ * S_ / 256, E_ / 64);   // (16, 16)
    gemm_bf16<0><<<gg, 256, 0, stream>>>(xb, wqt, bq, qb,  nullptr);
    gemm_bf16<0><<<gg, 256, 0, stream>>>(xb, wkt, bk, kb,  nullptr);
    gemm_bf16<2><<<gg, 256, 0, stream>>>(xb, wvt, bv, vtb, nullptr);

    attn_banded<<<B_ * H_ * (S_ / 16) / 4, 128, 0, stream>>>(qb, kb, vtb, aob);

    gemm_bf16<3><<<gg, 256, 0, stream>>>(aob, wot, bo, nullptr, out);
}
